// MultiBoxLoss_180388626931
// MI455X (gfx1250) — compile-verified, matrix operand order checked by reading, not run on hardware
//
#include <hip/hip_runtime.h>
#include <hip/hip_bf16.h>
#include <math.h>

typedef __attribute__((ext_vector_type(16))) _Float16 v16h;
typedef __attribute__((ext_vector_type(8)))  float    v8f;
typedef __attribute__((address_space(3))) float lds_f32;

#define BB    64
#define NOBJ  32
#define NPRI  8732
#define NCLS  81
#define THRESH 0.5f
#define SORTN 16384

// ---------------------------------------------------------------- init
__global__ void init_ws_kernel(unsigned long long* best, int* npos, float* acc) {
  int t = threadIdx.x;
  for (int i = t; i < BB * NOBJ; i += blockDim.x) best[i] = 0ull;
  if (t < BB) npos[t] = 0;
  if (t < 4)  acc[t]  = 0.0f;
}

// ------------------------------------------------------------- match
// ov_for_prior / obj_for_prior per (b,p); per-object best prior via packed
// 64-bit atomicMax: (iou_bits<<32)|(~p) -> lowest p wins on ties (first occurrence).
__global__ void match_kernel(const float* __restrict__ boxes,
                             const float* __restrict__ priors,
                             float* __restrict__ ovp,
                             int*   __restrict__ obj,
                             unsigned long long* __restrict__ best) {
  __shared__ float sbox[NOBJ * 4];
  __shared__ float sarea[NOBJ];
  __shared__ unsigned long long sbest[NOBJ];
  const int b   = blockIdx.y;
  const int tid = threadIdx.x;
  if (tid < NOBJ * 4) sbox[tid] = boxes[b * NOBJ * 4 + tid];
  __syncthreads();
  if (tid < NOBJ) {
    sarea[tid] = (sbox[tid * 4 + 2] - sbox[tid * 4 + 0]) *
                 (sbox[tid * 4 + 3] - sbox[tid * 4 + 1]);
    sbest[tid] = 0ull;
  }
  __syncthreads();
  const int p = blockIdx.x * blockDim.x + tid;
  if (p < NPRI) {
    float px1 = priors[p * 4 + 0], py1 = priors[p * 4 + 1];
    float px2 = priors[p * 4 + 2], py2 = priors[p * 4 + 3];
    float parea = (px2 - px1) * (py2 - py1);
    float bestv = -1.0f; int besto = 0;
    for (int o = 0; o < NOBJ; ++o) {
      float lx = fmaxf(sbox[o * 4 + 0], px1), ly = fmaxf(sbox[o * 4 + 1], py1);
      float rx = fminf(sbox[o * 4 + 2], px2), ry = fminf(sbox[o * 4 + 3], py2);
      float inter = fmaxf(rx - lx, 0.0f) * fmaxf(ry - ly, 0.0f);
      float iou = inter / (sarea[o] + parea - inter);
      if (iou > bestv) { bestv = iou; besto = o; }   // first occurrence on ties
      unsigned long long pk =
          ((unsigned long long)__float_as_uint(iou) << 32) |
          (unsigned long long)(0xFFFFFFFFu - (unsigned)p);
      atomicMax(&sbest[o], pk);
    }
    ovp[(size_t)b * NPRI + p] = bestv;
    obj[(size_t)b * NPRI + p] = besto;
  }
  __syncthreads();
  if (tid < NOBJ) atomicMax(&best[b * NOBJ + tid], sbest[tid]);
}

// --------------------------------------------------------- override
// Sequential per batch so duplicate prior indices resolve last-wins.
__global__ void override_kernel(const unsigned long long* __restrict__ best,
                                float* __restrict__ ovp, int* __restrict__ obj) {
  int b = blockIdx.x * blockDim.x + threadIdx.x;
  if (b >= BB) return;
  for (int j = 0; j < NOBJ; ++j) {
    unsigned p = 0xFFFFFFFFu - (unsigned)(best[b * NOBJ + j] & 0xFFFFFFFFull);
    obj[(size_t)b * NPRI + p] = j;
    ovp[(size_t)b * NPRI + p] = 1.0f;
  }
}

// --------------------------------------------- label assign + loc L1
__global__ void assign_loc_kernel(const float* __restrict__ plocs,
                                  const float* __restrict__ boxes,
                                  const int*   __restrict__ labels,
                                  const float* __restrict__ priors,
                                  const float* __restrict__ ovp,
                                  const int*   __restrict__ obj,
                                  int*   __restrict__ tc,
                                  int*   __restrict__ npos,
                                  float* __restrict__ acc) {
  __shared__ float sf[256];
  __shared__ int   si[256];
  const int b   = blockIdx.y;
  const int tid = threadIdx.x;
  const int p   = blockIdx.x * blockDim.x + tid;
  float l1 = 0.0f; int cnt = 0;
  if (p < NPRI) {
    size_t idx = (size_t)b * NPRI + p;
    int o = obj[idx];
    float ov = ovp[idx];
    int lab = labels[b * NOBJ + o];
    if (ov < THRESH) lab = 0;
    tc[idx] = lab;
    if (lab != 0) {
      cnt = 1;
      const float* bx = boxes + ((size_t)b * NOBJ + o) * 4;
      float x1 = bx[0], y1 = bx[1], x2 = bx[2], y2 = bx[3];
      float cx = (x1 + x2) * 0.5f, cy = (y1 + y2) * 0.5f;
      float w = x2 - x1, h = y2 - y1;
      float pcx = priors[p * 4 + 0], pcy = priors[p * 4 + 1];
      float pw  = priors[p * 4 + 2], ph  = priors[p * 4 + 3];
      float g0 = (cx - pcx) * 10.0f / pw;
      float g1 = (cy - pcy) * 10.0f / ph;
      float g2 = logf(w / pw) * 5.0f;
      float g3 = logf(h / ph) * 5.0f;
      const float* pl = plocs + idx * 4;
      l1 = fabsf(pl[0] - g0) + fabsf(pl[1] - g1) +
           fabsf(pl[2] - g2) + fabsf(pl[3] - g3);
    }
  }
  sf[tid] = l1; si[tid] = cnt;
  __syncthreads();
  for (int s = 128; s > 0; s >>= 1) {
    if (tid < s) { sf[tid] += sf[tid + s]; si[tid] += si[tid + s]; }
    __syncthreads();
  }
  if (tid == 0) { atomicAdd(&acc[0], sf[0]); atomicAdd(&npos[b], si[0]); }
}

// ----------------------------------------- cross entropy via WMMA
// One wave32 per 16-prior tile. A = exp(x - rowmax) (16x96 f16, ISA A-layout),
// B = ones -> D row sums in f32 accumulator via v_wmma_f32_16x16x32_f16.
__global__ void __launch_bounds__(128)
ce_wmma_kernel(const float* __restrict__ scores,
               const int*   __restrict__ tc,
               float* __restrict__ ce_neg,
               float* __restrict__ acc) {
  __shared__ float rowsum[4][16];
  const int b    = blockIdx.y;
  const int tid  = threadIdx.x;
  const int wave = tid >> 5;
  const int lane = tid & 31;
  const int tile = blockIdx.x * 4 + wave;
  const int pbase = tile * 16;
  if (pbase >= NPRI) return;                 // whole-wave uniform exit, no barriers used

  const int r  = lane & 15;                  // matrix row handled by this lane
  const int hi = lane >> 4;                  // K-half selector per ISA A layout
  int p = pbase + r; if (p >= NPRI) p = NPRI - 1;
  const float* row = scores + ((size_t)b * NPRI + p) * NCLS;

  { // prefetch 4 tiles ahead -> global_prefetch_b8
    int pn = pbase + 64 + r; if (pn >= NPRI) pn = NPRI - 1;
    __builtin_prefetch(scores + ((size_t)b * NPRI + pn) * NCLS, 0, 1);
  }

  // Load this lane's 48 class values (3 chunks x {8 + 8} per A layout), track max.
  float vals[48];
  float pm = -INFINITY;
#pragma unroll
  for (int c = 0; c < 3; ++c) {
#pragma unroll
    for (int seg = 0; seg < 2; ++seg) {
      int k0 = 32 * c + seg * 16 + 8 * hi;
#pragma unroll
      for (int i = 0; i < 8; ++i) {
        int k = k0 + i;
        float v = (k < NCLS) ? row[k] : -INFINITY;
        vals[c * 16 + seg * 8 + i] = v;
        pm = fmaxf(pm, v);
      }
    }
  }
  // Lanes l and l+16 jointly cover row l; combine partial maxima.
  float rowmax = fmaxf(pm, __shfl_xor(pm, 16, 32));

  v16h ones;
#pragma unroll
  for (int i = 0; i < 16; ++i) ones[i] = (_Float16)1.0f;

  v8f c8 = {};
#pragma unroll
  for (int c = 0; c < 3; ++c) {
    v16h a;
#pragma unroll
    for (int e = 0; e < 16; ++e)
      a[e] = (_Float16)__expf(vals[c * 16 + e] - rowmax);
    c8 = __builtin_amdgcn_wmma_f32_16x16x32_f16(
        false, a, false, ones, (short)0, c8, false, false);
  }

  // D column N=0: lane 0 holds rows 0..7 in c8[0..7], lane 16 holds rows 8..15.
  if (lane == 0) {
#pragma unroll
    for (int m = 0; m < 8; ++m) rowsum[wave][m] = c8[m];
  } else if (lane == 16) {
#pragma unroll
    for (int m = 0; m < 8; ++m) rowsum[wave][8 + m] = c8[m];
  }
  asm volatile("s_wait_dscnt 0" ::: "memory");   // DS in-order per wave; drain before read

  float contrib = 0.0f;
  if (hi == 0 && (pbase + r) < NPRI) {
    float s = rowsum[wave][r];
    size_t idx = (size_t)b * NPRI + (pbase + r);
    int lab = tc[idx];
    float xl = scores[idx * NCLS + lab];
    float ce = rowmax + __logf(s) - xl;        // -log_softmax[lab]
    if (lab != 0) { ce_neg[idx] = 0.0f; contrib = ce; }
    else          { ce_neg[idx] = ce; }
  }
  contrib += __shfl_xor(contrib, 16, 32);
  contrib += __shfl_xor(contrib,  8, 32);
  contrib += __shfl_xor(contrib,  4, 32);
  contrib += __shfl_xor(contrib,  2, 32);
  contrib += __shfl_xor(contrib,  1, 32);
  if (lane == 0) atomicAdd(&acc[1], contrib);
}

// -------------------------------------- hard negative mining (top-K sum)
// One block per batch; stage ce_neg[b] into LDS with gfx1250 async-to-LDS DMA
// (bypasses VGPRs, tracked on ASYNCcnt), then descending bitonic sort in LDS.
__global__ void hardneg_kernel(const float* __restrict__ ce_neg,
                               const int*   __restrict__ npos,
                               float* __restrict__ acc) {
  __shared__ float s[SORTN];
  const int b   = blockIdx.x;
  const int tid = threadIdx.x;
  const float* src = ce_neg + (size_t)b * NPRI;

  // Async global -> LDS staging (per-lane addresses). VDST = LDS byte offset,
  // ADDR = 64-bit global address ("GV" mode), counted on ASYNCcnt.
  for (int i = tid; i < NPRI; i += blockDim.x) {
    unsigned lds_off = (unsigned)(unsigned long long)(lds_f32*)&s[i];
    const float* g = src + i;
    asm volatile("global_load_async_to_lds_b32 %0, %1, off"
                 :: "v"(lds_off), "v"(g) : "memory");
  }
  for (int i = NPRI + tid; i < SORTN; i += blockDim.x)
    s[i] = -1.0f;                              // ce >= 0, pads sort last
  asm volatile("s_wait_asynccnt 0" ::: "memory");
  __syncthreads();

  for (int k = 2; k <= SORTN; k <<= 1) {
    for (int j = k >> 1; j > 0; j >>= 1) {
      for (int t = tid; t < SORTN; t += blockDim.x) {
        int ixj = t ^ j;
        if (ixj > t) {
          float a = s[t], c = s[ixj];
          bool up = (t & k) == 0;               // descending overall
          if (up ? (a < c) : (a > c)) { s[t] = c; s[ixj] = a; }
        }
      }
      __syncthreads();
    }
  }
  int K = 3 * npos[b]; if (K > NPRI) K = NPRI;
  float sum = 0.0f;
  for (int i = tid; i < K; i += blockDim.x) sum += s[i];
  __syncthreads();
  sum += __shfl_xor(sum, 16, 32);
  sum += __shfl_xor(sum,  8, 32);
  sum += __shfl_xor(sum,  4, 32);
  sum += __shfl_xor(sum,  2, 32);
  sum += __shfl_xor(sum,  1, 32);
  if ((tid & 31) == 0) s[16000 + (tid >> 5)] = sum;   // region past sorted K, safe
  __syncthreads();
  if (tid == 0) {
    float t = 0.0f;
    for (int w = 0; w < (int)(blockDim.x >> 5); ++w) t += s[16000 + w];
    atomicAdd(&acc[2], t);
  }
}

// --------------------------------------------------------- finalize
__global__ void finalize_kernel(const int* __restrict__ npos,
                                const float* __restrict__ acc,
                                float* __restrict__ out) {
  if (threadIdx.x == 0 && blockIdx.x == 0) {
    float npt = 0.0f;
    for (int b = 0; b < BB; ++b) npt += (float)npos[b];
    float conf = (acc[2] + acc[1]) / npt;
    float loc  = acc[0] / (npt * 4.0f);
    out[0] = conf + loc;
  }
}

// ------------------------------------------------------------ launch
extern "C" void kernel_launch(void* const* d_in, const int* in_sizes, int n_in,
                              void* d_out, int out_size, void* d_ws, size_t ws_size,
                              hipStream_t stream) {
  const float* plocs  = (const float*)d_in[0];   // [B, P, 4]
  const float* scores = (const float*)d_in[1];   // [B, P, 81]
  const float* boxes  = (const float*)d_in[2];   // [B, 32, 4]
  const int*   labels = (const int*)  d_in[3];   // [B, 32]
  const float* priors = (const float*)d_in[4];   // [P, 4]
  float* out = (float*)d_out;

  char* w = (char*)d_ws;
  unsigned long long* best = (unsigned long long*)w; w += (size_t)BB * NOBJ * 8;
  int*   npos   = (int*)w;   w += (size_t)BB * 4;
  float* acc    = (float*)w; w += 16;
  float* ovp    = (float*)w; w += (size_t)BB * NPRI * 4;
  int*   obj    = (int*)w;   w += (size_t)BB * NPRI * 4;
  int*   tc     = (int*)w;   w += (size_t)BB * NPRI * 4;
  float* ce_neg = (float*)w; w += (size_t)BB * NPRI * 4;

  init_ws_kernel<<<1, 256, 0, stream>>>(best, npos, acc);

  dim3 gMatch((NPRI + 255) / 256, BB);
  match_kernel<<<gMatch, 256, 0, stream>>>(boxes, priors, ovp, obj, best);

  override_kernel<<<1, 64, 0, stream>>>(best, ovp, obj);

  assign_loc_kernel<<<gMatch, 256, 0, stream>>>(plocs, boxes, labels, priors,
                                                ovp, obj, tc, npos, acc);

  int tiles = (NPRI + 15) / 16;                  // 546
  dim3 gCE((tiles + 3) / 4, BB);
  ce_wmma_kernel<<<gCE, 128, 0, stream>>>(scores, tc, ce_neg, acc);

  hardneg_kernel<<<BB, 256, 0, stream>>>(ce_neg, npos, acc);

  finalize_kernel<<<1, 1, 0, stream>>>(npos, acc, out);
}
